// EALSTM_798863917783
// MI455X (gfx1250) — matrix operand order repaired and checked by reading, not compiled, not run on hardware
//
#include <hip/hip_runtime.h>
#include <hip/hip_bf16.h>

// ---------------------------------------------------------------------------
// EA-LSTM for MI455X (gfx1250, wave32, WMMA)
//   B=256, T=365, D_dyn=32, D_stat=27, H=256, 3H=768
//
// v6: gate-aligned tile assignment -> elementwise directly on accumulators.
//   * Wave w owns N-tiles {w, w+16, w+32} = columns 16w..16w+16 of the
//     f / o / g gates respectively, so the LSTM update for cell (m, j)
//     uses acc0[r], acc1[r], acc2[r] in the same lane: NO gate staging
//     through LDS at all (deletes 48 DS ops/thread/step + 49 KB LDS).
//   * K-blocks 0..4 B-fragments pinned in registers; 5..8 streamed from L2
//     as global_load_b128 with immediate offsets via a laundered zero index.
//   * x_d prefetched one step ahead; outputs via uniform SGPR-stepped bases
//     + one voffset + immediate r*T*H offsets.
//   * tanh(x) = 2*sigmoid(2x) - 1 (v_exp based, no ocml branches).
// ---------------------------------------------------------------------------

typedef __attribute__((ext_vector_type(16))) _Float16 v16h;
typedef __attribute__((ext_vector_type(8)))  _Float16 half8;
typedef __attribute__((ext_vector_type(8)))  float    v8f;

#define BB     256
#define TT     365
#define DDYN   32
#define HH     256
#define G3     768        // 3*H
#define NKB    9          // K blocks of 32 (K = 256 + 32)
#define KB_REG 5          // K-blocks pinned in registers
#define NNT    48         // N tiles of 16
#define BT     16         // batch tile rows per workgroup
#define NWG    (BB / BT)  // 16 workgroups
#define THREADS 512       // 16 waves

#define ASTR   296        // A-lds row stride in halves (148 DW, conflict-free)

#define WSW_HALVES (NKB * NNT * 32 * 16)        // 221184 halves
#define WSW_BYTES  (WSW_HALVES * 2)             // 442368 B

// --------------------------- setup kernel 1 --------------------------------
// Swizzle concatenated weights into WMMA B-fragment order:
//   wsw[((kb*NNT + nt)*32 + lane)*16 + e]  holds  W[k][n]  f16
//   with n = nt*16 + (lane&15), k = kb*32 + (lane<16 ? 0 : 8) + (e&7) + (e&8 ? 16 : 0)
__global__ void ealstm_build_wswz(const float* __restrict__ Whh,
                                  const float* __restrict__ Wih,
                                  _Float16* __restrict__ wsw) {
    int idx = blockIdx.x * 256 + threadIdx.x;
    if (idx >= WSW_HALVES) return;
    int e  = idx & 15;
    int l  = (idx >> 4) & 31;
    int nt = (idx >> 9) % NNT;
    int kb = idx / (512 * NNT);
    int n  = nt * 16 + (l & 15);
    int k  = kb * 32 + ((l < 16) ? 0 : 8) + (e & 7) + ((e & 8) ? 16 : 0);
    float v = (k < HH) ? Whh[k * G3 + n] : Wih[(k - HH) * G3 + n];
    wsw[idx] = (_Float16)v;
}

// --------------------------- setup kernel 2 --------------------------------
__global__ void ealstm_static_gate(const float* __restrict__ xs,
                                   const float* __restrict__ Wsh,
                                   const float* __restrict__ bias_s,
                                   float* __restrict__ ig) {
    int b = blockIdx.x;
    int j = threadIdx.x;
    float s = bias_s[j];
#pragma unroll
    for (int k = 0; k < 27; ++k)
        s += xs[b * 27 + k] * Wsh[k * HH + j];
    ig[b * HH + j] = 1.0f / (1.0f + __expf(-s));
}

// --------------------------- recurrence kernel -----------------------------
__device__ __forceinline__ float sigm(float x) {
    return 1.0f / (1.0f + __expf(-x));
}
__device__ __forceinline__ float fast_tanh(float x) {
    return __builtin_fmaf(2.0f, sigm(2.0f * x), -1.0f);   // == tanh(x)
}

__device__ __forceinline__ v16h load_a_frag(const _Float16* Alds, int lane,
                                            int klo, int kb) {
    const half8* pa = (const half8*)&Alds[(lane & 15) * ASTR + kb * 32 + klo];
    half8 alo = pa[0];                 // K = base + 0..7
    half8 ahi = pa[2];                 // K = base + 16..23
    return __builtin_shufflevector(alo, ahi,
           0, 1, 2, 3, 4, 5, 6, 7, 8, 9, 10, 11, 12, 13, 14, 15);
}

__global__ void __launch_bounds__(THREADS, 1)
ealstm_recurrence(const float* __restrict__ xd,        // [B, T, 32]
                  const float* __restrict__ bias,      // [768]
                  const _Float16* __restrict__ wsw,    // swizzled weights
                  const float* __restrict__ ig,        // [B, 256] static gate
                  float* __restrict__ out)             // [2, B, T, 256]
{
    __shared__ _Float16 Alds[BT * ASTR];   // [16 x 296] halves: h | x_d(t) | pad

    const int tid  = threadIdx.x;
    const int lane = tid & 31;
    const int wv   = tid >> 5;             // 0..15
    const int b0   = blockIdx.x * BT;      // global batch base of this tile
    const size_t BTH = (size_t)BB * TT * HH;

    const int m_base = (lane < 16) ? 0 : 8;    // C/D fragment M offset
    const int n_loc  = lane & 15;              // C/D fragment N within tile
    const int klo    = (lane < 16) ? 0 : 8;    // A fragment K sub-offset
    const int jcol   = wv * 16 + n_loc;        // this thread's gate column j

    // ---- zero the h region of A-lds (any thread->cell mapping works) ----
#pragma unroll
    for (int e = 0; e < 8; ++e) {
        int idx = tid + e * THREADS;
        Alds[(idx >> 8) * ASTR + (idx & 255)] = (_Float16)0.0f;
    }

    // per-thread cell state in fragment layout: r -> (m = m_base+r, jcol)
    float c_reg[8], i_reg[8];
#pragma unroll
    for (int r = 0; r < 8; ++r) {
        c_reg[r] = 0.0f;
        i_reg[r] = ig[(b0 + m_base + r) * HH + jcol];
    }

    // bias folded into accumulator init (gate-aligned tiles!)
    const float bias0 = bias[jcol];            // f gate
    const float bias1 = bias[HH + jcol];       // o gate
    const float bias2 = bias[2 * HH + jcol];   // g gate

    // per-wave B-fragment base; tiles {wv, wv+16, wv+32} => v16h offsets
    // kb*1536 + {0, 512, 1024} from ps (pure immediates).
    const v16h* ps = ((const v16h*)wsw) + wv * 32 + lane;

    // ---- pin K-blocks 0..4 B-fragments in registers (120 VGPRs) ----
    v16h bfrag[3 * KB_REG];
#pragma unroll
    for (int kb = 0; kb < KB_REG; ++kb) {
        bfrag[3 * kb + 0] = ps[kb * NNT * 32 +    0];   // f tile (nt = wv)
        bfrag[3 * kb + 1] = ps[kb * NNT * 32 +  512];   // o tile (nt = wv+16)
        bfrag[3 * kb + 2] = ps[kb * NNT * 32 + 1024];   // g tile (nt = wv+32)
    }

    // x_d staging: per-thread invariant voffset + uniform stepping pointer
    const int xk  = tid & 31;
    const int xbr = tid >> 5;                  // 0..15
    const int xoff = (b0 + xbr) * TT * DDYN + xk;
    const int xlds = xbr * ASTR + HH + xk;
    const float* xptr = xd;                    // uniform, += DDYN per step

    // outputs: uniform stepping bases + ONE voffset + immediate r*T*H
    float* ohp = out;                          // h_n stream, += HH per step
    float* ocp = out + BTH;                    // c_n stream, += HH per step
    const int obase = (b0 + m_base) * TT * HH + jcol;
    const int hbase = m_base * ASTR + jcol;    // h write base in A-lds

    float xv = xptr[xoff];                     // prefetch t = 0

    for (int t = 0; t < TT; ++t) {
        // ---- stage this step's x element; prefetch next step's ----
        Alds[xlds] = (_Float16)xv;
        xptr += DDYN;
        float xnext = 0.0f;
        if (t < TT - 1) xnext = xptr[xoff];    // latency hidden behind step
        __syncthreads();   // h(t-1) + x(t) visible to all A-fragment reads

        v8f acc0 = {bias0, bias0, bias0, bias0, bias0, bias0, bias0, bias0};
        v8f acc1 = {bias1, bias1, bias1, bias1, bias1, bias1, bias1, bias1};
        v8f acc2 = {bias2, bias2, bias2, bias2, bias2, bias2, bias2, bias2};

        // ---- K-blocks 0..4 from register-pinned fragments ----
#pragma unroll
        for (int kb = 0; kb < KB_REG; ++kb) {
            v16h a = load_a_frag(Alds, lane, klo, kb);
            acc0 = __builtin_amdgcn_wmma_f32_16x16x32_f16(
                       false, a, false, bfrag[3 * kb + 0], (short)0, acc0, false, false);
            acc1 = __builtin_amdgcn_wmma_f32_16x16x32_f16(
                       false, a, false, bfrag[3 * kb + 1], (short)0, acc1, false, false);
            acc2 = __builtin_amdgcn_wmma_f32_16x16x32_f16(
                       false, a, false, bfrag[3 * kb + 2], (short)0, acc2, false, false);
        }

        // ---- K-blocks 5..8 streamed from L2 (global, non-hoistable) ----
        int zero = 0;
        asm volatile("" : "+v"(zero));
        const v16h* p = ps + zero;
#pragma unroll
        for (int kb = KB_REG; kb < NKB; ++kb) {
            v16h s0 = p[kb * NNT * 32 +    0];
            v16h s1 = p[kb * NNT * 32 +  512];
            v16h s2 = p[kb * NNT * 32 + 1024];
            v16h a  = load_a_frag(Alds, lane, klo, kb);
            acc0 = __builtin_amdgcn_wmma_f32_16x16x32_f16(
                       false, a, false, s0, (short)0, acc0, false, false);
            acc1 = __builtin_amdgcn_wmma_f32_16x16x32_f16(
                       false, a, false, s1, (short)0, acc1, false, false);
            acc2 = __builtin_amdgcn_wmma_f32_16x16x32_f16(
                       false, a, false, s2, (short)0, acc2, false, false);
            __builtin_amdgcn_sched_barrier(0);
        }

        __syncthreads();   // all A-fragment reads done; safe to overwrite h

        // ---- LSTM update directly on accumulator registers ----
        //   acc0[r]=f, acc1[r]=o, acc2[r]=g for cell (m_base+r, jcol)
#pragma unroll
        for (int r = 0; r < 8; ++r) {
            float c1 = sigm(acc0[r]) * c_reg[r] + i_reg[r] * fast_tanh(acc2[r]);
            float h1 = sigm(acc1[r]) * fast_tanh(c1);
            c_reg[r] = c1;
            ohp[obase + r * (TT * HH)] = h1;   // h_n (immediate r offsets)
            ocp[obase + r * (TT * HH)] = c1;   // c_n
            Alds[hbase + r * ASTR] = (_Float16)h1;   // h for next step
        }
        ohp += HH;
        ocp += HH;
        xv = xnext;
        // next iteration's top barrier orders h-writes vs. the next matmul
    }
}

// ------------------------------- launcher ----------------------------------
extern "C" void kernel_launch(void* const* d_in, const int* in_sizes, int n_in,
                              void* d_out, int out_size, void* d_ws, size_t ws_size,
                              hipStream_t stream) {
    (void)in_sizes; (void)n_in; (void)out_size; (void)ws_size;
    const float* x_d    = (const float*)d_in[0];  // [B, T, 32]
    const float* x_s    = (const float*)d_in[1];  // [B, 27]
    const float* w_ih   = (const float*)d_in[2];  // [32, 768]
    const float* w_hh   = (const float*)d_in[3];  // [256, 768]
    const float* w_sh   = (const float*)d_in[4];  // [27, 256]
    const float* bias   = (const float*)d_in[5];  // [768]
    const float* bias_s = (const float*)d_in[6];  // [256]
    float* out = (float*)d_out;

    // workspace: [0, 442368) swizzled f16 weights; then [B*H] f32 static gate
    _Float16* wsw = (_Float16*)d_ws;
    float*    ig  = (float*)((char*)d_ws + WSW_BYTES);

    ealstm_build_wswz<<<(WSW_HALVES + 255) / 256, 256, 0, stream>>>(w_hh, w_ih, wsw);
    ealstm_static_gate<<<BB, HH, 0, stream>>>(x_s, w_sh, bias_s, ig);
    ealstm_recurrence<<<NWG, THREADS, 0, stream>>>(x_d, bias, wsw, ig, out);
}
